// deformableLKABlock_11003706212446
// MI455X (gfx1250) — matrix operand order repaired and checked by reading, not compiled, not run on hardware
//
#include <hip/hip_runtime.h>
#include <hip/hip_bf16.h>
#include <math.h>

// Problem constants
#define BB   4
#define CCH  96
#define HH   56
#define WW   56
#define HW   3136            // 56*56
#define PIX  12544           // BB*HW
#define HID  4

typedef __attribute__((ext_vector_type(16))) _Float16 v16h;
typedef __attribute__((ext_vector_type(8)))  float    v8f;
typedef __attribute__((ext_vector_type(4)))  int      v4i_b128;

union HF { v16h v; uint32_t u[8]; _Float16 h[16]; };

// --- CDNA5 async global->LDS path (guarded: falls back to plain LDS stores) ---
#if defined(__has_builtin)
#if __has_builtin(__builtin_amdgcn_global_load_async_to_lds_b128) && \
    __has_builtin(__builtin_amdgcn_s_wait_asynccnt)
#define USE_ASYNC_LDS 1
#endif
#endif
#ifndef USE_ASYNC_LDS
#define USE_ASYNC_LDS 0
#endif

#define AS_GLOBAL __attribute__((address_space(1)))
#define AS_LDS    __attribute__((address_space(3)))

__device__ __forceinline__ float gelu_exact(float v) {
    return 0.5f * v * (1.0f + erff(v * 0.7071067811865476f));
}

__device__ __forceinline__ v16h frag_zero() {
    HF r;
    #pragma unroll
    for (int i = 0; i < 8; ++i) r.u[i] = 0u;
    return r.v;
}

// A fragment: act row-major [rows x 96] f16.
// A 16x32 layout: lane(0-15)=M, halves j0..15 -> K=j+16*hi (contiguous, matches B)
__device__ __forceinline__ v16h frag_A(const _Float16* __restrict__ act, int row, int kb, int hi) {
    HF r;
    const uint32_t* p = (const uint32_t*)(act + (size_t)row * CCH + kb + hi * 16);
    r.u[0] = p[0]; r.u[1] = p[1]; r.u[2] = p[2];  r.u[3] = p[3];
    r.u[4] = p[4]; r.u[5] = p[5]; r.u[6] = p[6]; r.u[7] = p[7];
    return r.v;
}

// B fragment from LDS-staged [16 x 96] tile: lane&15 = N row, halves contiguous in K.
__device__ __forceinline__ v16h frag_B_lds(const _Float16* wlds, int l16, int kb, int hi) {
    HF r;
    const uint32_t* p = (const uint32_t*)(wlds + l16 * CCH + kb + hi * 16);
    #pragma unroll
    for (int i = 0; i < 8; ++i) r.u[i] = p[i];
    return r.v;
}

// Stage one 16x96 f16 weight tile into LDS (rows >= Nvalid zero-filled).
// 192 threads move one 16B chunk each; async global->LDS when available.
__device__ __forceinline__ void stage_wtile(_Float16* wlds,
                                            const _Float16* __restrict__ wsrc,
                                            int gn_base, int Nvalid) {
    int t = threadIdx.x;
    if (t < 192) {
        int nrow = t / 12, cidx = t % 12;
        _Float16* dst = wlds + nrow * CCH + cidx * 8;
        int gn = gn_base + nrow;
        if (gn < Nvalid) {
            const _Float16* src = wsrc + (size_t)gn * CCH + cidx * 8;
#if USE_ASYNC_LDS
            __builtin_amdgcn_global_load_async_to_lds_b128(
                (AS_GLOBAL v4i_b128*)src,
                (AS_LDS    v4i_b128*)dst, 0, 0);
#else
            *(int4*)dst = *(const int4*)src;
#endif
        } else {
            int4 z; z.x = 0; z.y = 0; z.z = 0; z.w = 0;
            *(int4*)dst = z;
        }
    }
#if USE_ASYNC_LDS
    __builtin_amdgcn_s_wait_asynccnt(0);
#endif
    __syncthreads();
}

// ---------------- weight cast kernels ----------------
__global__ void cast_h_kernel(const float* __restrict__ in, _Float16* __restrict__ out, int n) {
    int i = blockIdx.x * blockDim.x + threadIdx.x;
    if (i < n) out[i] = (_Float16)in[i];
}

// (O,C,K,K) f32 -> [tap][O][C] f16 for contiguous B loads in implicit-GEMM conv
__global__ void cast_convw_kernel(const float* __restrict__ in, _Float16* __restrict__ out,
                                  int O, int KK2) {
    int i = blockIdx.x * blockDim.x + threadIdx.x;
    int n = O * CCH * KK2;
    if (i >= n) return;
    int kidx = i % KK2;
    int c = (i / KK2) % CCH;
    int o = i / (KK2 * CCH);
    out[((size_t)kidx * O + o) * CCH + c] = (_Float16)in[i];
}

// ---------------- LayerNorm over channels ----------------
__global__ void ln_nchw_kernel(const float* __restrict__ x,
                               const float* __restrict__ g, const float* __restrict__ bta,
                               float* __restrict__ out_f, _Float16* __restrict__ out_h) {
    int wid = threadIdx.x >> 5, lane = threadIdx.x & 31;
    int p = blockIdx.x * (blockDim.x >> 5) + wid;
    int bimg = p / HW, hwp = p % HW;
    const float* base = x + (size_t)bimg * CCH * HW + hwp;
    float v[3];
    #pragma unroll
    for (int i = 0; i < 3; ++i) v[i] = base[(size_t)(lane + 32 * i) * HW];
    float s = v[0] + v[1] + v[2];
    float ss = v[0]*v[0] + v[1]*v[1] + v[2]*v[2];
    #pragma unroll
    for (int m = 16; m > 0; m >>= 1) { s += __shfl_xor(s, m, 32); ss += __shfl_xor(ss, m, 32); }
    float mean = s * (1.0f / 96.0f);
    float var = ss * (1.0f / 96.0f) - mean * mean;
    float rs = rsqrtf(var + 1e-5f);
    #pragma unroll
    for (int i = 0; i < 3; ++i) {
        int c = lane + 32 * i;
        float o = (v[i] - mean) * rs * g[c] + bta[c];
        out_f[(size_t)p * CCH + c] = o;
        out_h[(size_t)p * CCH + c] = (_Float16)o;
    }
}

__global__ void ln_pc_kernel(const float* __restrict__ in,
                             const float* __restrict__ g, const float* __restrict__ bta,
                             _Float16* __restrict__ out_h) {
    int wid = threadIdx.x >> 5, lane = threadIdx.x & 31;
    int p = blockIdx.x * (blockDim.x >> 5) + wid;
    const float* base = in + (size_t)p * CCH;
    float v[3];
    #pragma unroll
    for (int i = 0; i < 3; ++i) v[i] = base[lane + 32 * i];
    float s = v[0] + v[1] + v[2];
    float ss = v[0]*v[0] + v[1]*v[1] + v[2]*v[2];
    #pragma unroll
    for (int m = 16; m > 0; m >>= 1) { s += __shfl_xor(s, m, 32); ss += __shfl_xor(ss, m, 32); }
    float mean = s * (1.0f / 96.0f);
    float var = ss * (1.0f / 96.0f) - mean * mean;
    float rs = rsqrtf(var + 1e-5f);
    #pragma unroll
    for (int i = 0; i < 3; ++i) {
        int c = lane + 32 * i;
        out_h[(size_t)p * CCH + c] = (_Float16)((v[i] - mean) * rs * g[c] + bta[c]);
    }
}

// ---------------- 96-K GEMM (1x1 convs), fused epilogues ----------------
#define EPI_GELU_BOTH 0
#define EPI_MUL_U     1
#define EPI_LS_ADD    2
#define EPI_PLAIN     3

// Block = 8 waves; all waves share one N-tile (weights staged in LDS once),
// wave w handles M-tile bm*8+w. Grid = (784/8) * tiles_n blocks.
__global__ void gemm96_kernel(const _Float16* __restrict__ act,
                              const _Float16* __restrict__ w,
                              const float* __restrict__ bias,
                              int Nout, int tiles_n, int mode,
                              const float* __restrict__ extra,   // u (mode1) or shortcut (mode2)
                              const float* __restrict__ scale,   // ls1 (mode2)
                              float* __restrict__ out_f,
                              _Float16* __restrict__ out_h) {
    __shared__ __align__(16) _Float16 wlds[16 * CCH];
    const int bm = blockIdx.x / tiles_n;
    const int tn = blockIdx.x % tiles_n;
    const int wid = threadIdx.x >> 5;
    const int lane = threadIdx.x & 31;
    const int tm = bm * 8 + wid;
    const int hi = lane >> 4, l16 = lane & 15;
    const int row = tm * 16 + l16;      // A: lane = M

    stage_wtile(wlds, w, tn * 16, Nout);

    __builtin_prefetch(act + (size_t)row * CCH, 0, 0);   // global_prefetch_b8

    v8f acc = {};
    #pragma unroll
    for (int kb = 0; kb < CCH; kb += 32) {
        v16h a = frag_A(act, row, kb, hi);
        v16h b = frag_B_lds(wlds, l16, kb, hi);
        acc = __builtin_amdgcn_wmma_f32_16x16x32_f16(false, a, false, b, (short)0, acc, false, false);
    }

    const int col = tn * 16 + l16;      // C: lane = N, vgpr r -> M = r + 8*hi
    #pragma unroll
    for (int r = 0; r < 8; ++r) {
        int m = r + hi * 8;
        int orow = tm * 16 + m;
        if (col < Nout) {
            float v = acc[r] + bias[col];
            size_t idx = (size_t)orow * Nout + col;
            if (mode == EPI_GELU_BOTH) {
                float gv = gelu_exact(v);
                out_f[idx] = gv;
                out_h[idx] = (_Float16)gv;
            } else if (mode == EPI_MUL_U) {
                out_h[idx] = (_Float16)(v * extra[idx]);
            } else if (mode == EPI_LS_ADD) {
                out_f[idx] = scale[col] * (v + extra[idx]);
            } else {
                out_f[idx] = v;
            }
        }
    }
}

// ---------------- implicit-GEMM KxK (dilated) conv for offset generation ----------------
__global__ void gemmconv_kernel(const _Float16* __restrict__ act,   // [P x 96]
                                const _Float16* __restrict__ wt,    // [KK2][O][96]
                                const float* __restrict__ bias,
                                int O, int tiles_n, int Kk, int dil, int pad,
                                float* __restrict__ out) {          // [P x O]
    __shared__ __align__(16) _Float16 wlds[16 * CCH];
    const int bm = blockIdx.x / tiles_n;
    const int tn = blockIdx.x % tiles_n;
    const int wid = threadIdx.x >> 5;
    const int lane = threadIdx.x & 31;
    const int tm = bm * 8 + wid;
    const int hi = lane >> 4, l16 = lane & 15;
    const int prow = tm * 16 + l16;
    const int bimg = prow / HW, hwp = prow % HW;
    const int hcur = hwp / WW, wcur = hwp % WW;
    const int KK2 = Kk * Kk;

    v8f acc = {};
    for (int kidx = 0; kidx < KK2; ++kidx) {
        // stage this tap's weight tile into LDS (shared by all 8 waves)
        stage_wtile(wlds, wt + (size_t)kidx * O * CCH, tn * 16, O);

        int dy = (kidx / Kk) * dil - pad;
        int dx = (kidx % Kk) * dil - pad;
        int sh = hcur + dy, sw = wcur + dx;
        bool valid = (sh >= 0) && (sh < HH) && (sw >= 0) && (sw < WW);
        int srow = bimg * HW + sh * WW + sw;
        #pragma unroll
        for (int kb = 0; kb < CCH; kb += 32) {
            v16h a = valid ? frag_A(act, srow, kb, hi) : frag_zero();
            v16h b = frag_B_lds(wlds, l16, kb, hi);
            acc = __builtin_amdgcn_wmma_f32_16x16x32_f16(false, a, false, b, (short)0, acc, false, false);
        }
        __syncthreads();   // protect LDS tile before next tap's staging
    }

    const int col = tn * 16 + l16;
    #pragma unroll
    for (int r = 0; r < 8; ++r) {
        int m = r + hi * 8;
        int orow = tm * 16 + m;
        if (col < O) out[(size_t)orow * O + col] = acc[r] + bias[col];
    }
}

// ---------------- deformable depthwise gather (bilinear) ----------------
__device__ __forceinline__ float dsample(const float* __restrict__ src, int b, int y, int x, int c) {
    if (y < 0 || y >= HH || x < 0 || x >= WW) return 0.0f;
    return src[((size_t)(b * HW + y * WW + x)) * CCH + c];
}

__global__ void deform_kernel(const float* __restrict__ src,   // [P x 96]
                              const float* __restrict__ off,   // [P x 2*KK2]
                              const float* __restrict__ dw,    // [96 x KK2]
                              const float* __restrict__ db,
                              int Kk, int dil, int pad,
                              float* __restrict__ out_f,
                              _Float16* __restrict__ out_h) {
    const int p = blockIdx.x;
    const int c = threadIdx.x;            // 96 channel threads
    const int bimg = p / HW, hwp = p % HW;
    const int hcur = hwp / WW, wcur = hwp % WW;
    const int KK2 = Kk * Kk;
    const float* ob = off + (size_t)p * 2 * KK2;

    float sum = 0.0f;
    for (int k = 0; k < KK2; ++k) {
        float py = (float)(hcur + (k / Kk) * dil - pad) + ob[2 * k];
        float px = (float)(wcur + (k % Kk) * dil - pad) + ob[2 * k + 1];
        float y0f = floorf(py), x0f = floorf(px);
        float wy = py - y0f, wx = px - x0f;
        int y0 = (int)y0f, x0 = (int)x0f;
        float v00 = dsample(src, bimg, y0,     x0,     c);
        float v01 = dsample(src, bimg, y0,     x0 + 1, c);
        float v10 = dsample(src, bimg, y0 + 1, x0,     c);
        float v11 = dsample(src, bimg, y0 + 1, x0 + 1, c);
        float bil = v00 * (1.0f - wy) * (1.0f - wx) + v01 * (1.0f - wy) * wx
                  + v10 * wy * (1.0f - wx)          + v11 * wy * wx;
        sum += dw[(size_t)c * KK2 + k] * bil;
    }
    float v = sum + db[c];
    out_f[(size_t)p * CCH + c] = v;
    out_h[(size_t)p * CCH + c] = (_Float16)v;
}

// ---------------- depthwise 3x3 on HID channels + GELU ----------------
__global__ void dwconv_kernel(const float* __restrict__ z1,   // [P x 4]
                              const float* __restrict__ dww,  // [4][1][3][3]
                              const float* __restrict__ dwb,
                              float* __restrict__ z2) {
    int tid = blockIdx.x * blockDim.x + threadIdx.x;
    if (tid >= PIX * HID) return;
    int p = tid >> 2, c = tid & 3;
    int bimg = p / HW, hwp = p % HW, h = hwp / WW, w = hwp % WW;
    float sum = 0.0f;
    #pragma unroll
    for (int ky = 0; ky < 3; ++ky)
        #pragma unroll
        for (int kx = 0; kx < 3; ++kx) {
            int sh = h + ky - 1, sw = w + kx - 1;
            if (sh >= 0 && sh < HH && sw >= 0 && sw < WW)
                sum += z1[((size_t)(bimg * HW + sh * WW + sw)) * HID + c] * dww[c * 9 + ky * 3 + kx];
        }
    z2[tid] = gelu_exact(sum + dwb[c]);
}

// ---------------- final: f2 (HID->C dot), layer-scale 2, residuals, NCHW out ----------------
__global__ void final_kernel(const float* __restrict__ x,     // NCHW
                             const float* __restrict__ ybr,   // [P x 96]
                             const float* __restrict__ z2,    // [P x 4]
                             const float* __restrict__ f2w,   // [96 x 4]
                             const float* __restrict__ f2b,
                             const float* __restrict__ ls2,
                             float* __restrict__ out) {
    int tid = blockIdx.x * blockDim.x + threadIdx.x;
    if (tid >= BB * CCH * HW) return;
    int hw = tid % HW;
    int c = (tid / HW) % CCH;
    int b = tid / (HW * CCH);
    int p = b * HW + hw;
    float z3 = f2b[c];
    #pragma unroll
    for (int j = 0; j < HID; ++j) z3 += z2[(size_t)p * HID + j] * f2w[c * HID + j];
    out[tid] = x[tid] + ybr[(size_t)p * CCH + c] + ls2[c] * z3;
}

// ---------------- host launcher ----------------
extern "C" void kernel_launch(void* const* d_in, const int* in_sizes, int n_in,
                              void* d_out, int out_size, void* d_ws, size_t ws_size,
                              hipStream_t stream) {
    (void)in_sizes; (void)n_in; (void)out_size; (void)ws_size;
    const float* x   = (const float*)d_in[0];
    const float* g1n = (const float*)d_in[1];
    const float* b1n = (const float*)d_in[2];
    const float* p1w = (const float*)d_in[3];
    const float* p1b = (const float*)d_in[4];
    const float* o0w = (const float*)d_in[5];
    const float* o0b = (const float*)d_in[6];
    const float* d0w = (const float*)d_in[7];
    const float* d0b = (const float*)d_in[8];
    const float* o1w = (const float*)d_in[9];
    const float* o1b = (const float*)d_in[10];
    const float* d1w = (const float*)d_in[11];
    const float* d1b = (const float*)d_in[12];
    const float* c1w = (const float*)d_in[13];
    const float* c1b = (const float*)d_in[14];
    const float* p2w = (const float*)d_in[15];
    const float* p2b = (const float*)d_in[16];
    const float* ls1 = (const float*)d_in[17];
    const float* g2n = (const float*)d_in[18];
    const float* b2n = (const float*)d_in[19];
    const float* f1w = (const float*)d_in[20];
    const float* f1b = (const float*)d_in[21];
    const float* dww = (const float*)d_in[22];
    const float* dwb = (const float*)d_in[23];
    const float* f2w = (const float*)d_in[24];
    const float* f2b = (const float*)d_in[25];
    const float* ls2 = (const float*)d_in[26];
    float* out = (float*)d_out;

    // workspace carve-out (256B aligned)
    char* base = (char*)d_ws;
    size_t woff = 0;
    auto alloc = [&](size_t bytes) -> void* {
        void* p = base + woff;
        woff = (woff + bytes + 255) & ~(size_t)255;
        return p;
    };
    const size_t PC = (size_t)PIX * CCH;
    float*     yln_f = (float*)alloc(PC * 4);
    _Float16*  yln_h = (_Float16*)alloc(PC * 2);
    float*     y1_f  = (float*)alloc(PC * 4);
    _Float16*  y1_h  = (_Float16*)alloc(PC * 2);
    float*     a0_f  = (float*)alloc(PC * 4);
    _Float16*  a0_h  = (_Float16*)alloc(PC * 2);
    float*     a1_f  = (float*)alloc(PC * 4);
    _Float16*  a1_h  = (_Float16*)alloc(PC * 2);
    _Float16*  m_h   = (_Float16*)alloc(PC * 2);
    float*     ybr_f = (float*)alloc(PC * 4);
    _Float16*  z_h   = (_Float16*)alloc(PC * 2);
    float*     z1    = (float*)alloc((size_t)PIX * HID * 4);
    float*     z2    = (float*)alloc((size_t)PIX * HID * 4);
    float*     off0  = (float*)alloc((size_t)PIX * 50 * 4);
    float*     off1  = (float*)alloc((size_t)PIX * 98 * 4);
    _Float16*  p1wh  = (_Float16*)alloc(9216 * 2);
    _Float16*  c1wh  = (_Float16*)alloc(9216 * 2);
    _Float16*  p2wh  = (_Float16*)alloc(9216 * 2);
    _Float16*  f1wh  = (_Float16*)alloc(HID * CCH * 2);
    _Float16*  o0wt  = (_Float16*)alloc((size_t)50 * CCH * 25 * 2);
    _Float16*  o1wt  = (_Float16*)alloc((size_t)98 * CCH * 49 * 2);

    // weight casts / repacks
    cast_h_kernel<<<(9216 + 255) / 256, 256, 0, stream>>>(p1w, p1wh, 9216);
    cast_h_kernel<<<(9216 + 255) / 256, 256, 0, stream>>>(c1w, c1wh, 9216);
    cast_h_kernel<<<(9216 + 255) / 256, 256, 0, stream>>>(p2w, p2wh, 9216);
    cast_h_kernel<<<(HID * CCH + 255) / 256, 256, 0, stream>>>(f1w, f1wh, HID * CCH);
    cast_convw_kernel<<<(50 * CCH * 25 + 255) / 256, 256, 0, stream>>>(o0w, o0wt, 50, 25);
    cast_convw_kernel<<<(98 * CCH * 49 + 255) / 256, 256, 0, stream>>>(o1w, o1wt, 98, 49);

    // attention branch
    ln_nchw_kernel<<<PIX / 8, 256, 0, stream>>>(x, g1n, b1n, yln_f, yln_h);
    gemm96_kernel<<<98 * 6, 256, 0, stream>>>(yln_h, p1wh, p1b, 96, 6, EPI_GELU_BOTH,
                                              nullptr, nullptr, y1_f, y1_h);
    gemmconv_kernel<<<98 * 4, 256, 0, stream>>>(y1_h, o0wt, o0b, 50, 4, 5, 1, 2, off0);
    deform_kernel<<<PIX, CCH, 0, stream>>>(y1_f, off0, d0w, d0b, 5, 1, 2, a0_f, a0_h);
    gemmconv_kernel<<<98 * 7, 256, 0, stream>>>(a0_h, o1wt, o1b, 98, 7, 7, 3, 9, off1);
    deform_kernel<<<PIX, CCH, 0, stream>>>(a0_f, off1, d1w, d1b, 7, 3, 9, a1_f, a1_h);
    gemm96_kernel<<<98 * 6, 256, 0, stream>>>(a1_h, c1wh, c1b, 96, 6, EPI_MUL_U,
                                              y1_f, nullptr, nullptr, m_h);
    gemm96_kernel<<<98 * 6, 256, 0, stream>>>(m_h, p2wh, p2b, 96, 6, EPI_LS_ADD,
                                              yln_f, ls1, ybr_f, nullptr);

    // MLP branch
    ln_pc_kernel<<<PIX / 8, 256, 0, stream>>>(ybr_f, g2n, b2n, z_h);
    gemm96_kernel<<<98, 256, 0, stream>>>(z_h, f1wh, f1b, HID, 1, EPI_PLAIN,
                                          nullptr, nullptr, z1, nullptr);
    dwconv_kernel<<<(PIX * HID + 255) / 256, 256, 0, stream>>>(z1, dww, dwb, z2);
    final_kernel<<<(PIX * CCH + 255) / 256, 256, 0, stream>>>(x, ybr_f, z2, f2w, f2b, ls2, out);
}
